// SwitchFeedForward_46402826666528
// MI455X (gfx1250) — matrix operand order repaired and checked by reading, not compiled, not run on hardware
//
#include <hip/hip_runtime.h>
#include <hip/hip_bf16.h>
#include <stdint.h>

// ---------------- problem constants (match reference) ----------------
#define T_TOK   4096        // S*B
#define DMODEL  768
#define FDIM    3072
#define NEXP    8
#define BASE_CAP 512        // int(1.0 * T/E)
#define CAPMAX   768        // C = base_cap + base_cap/2 (static buffer rows/expert)

typedef __attribute__((ext_vector_type(16))) __bf16 v16bf;
typedef __attribute__((ext_vector_type(8)))  __bf16 v8bf;
typedef __attribute__((ext_vector_type(8)))  float  v8f;

// Branchless round-to-nearest-even f32 -> bf16 (NaN quieted via select, no EXEC branches)
__device__ __forceinline__ unsigned short f32_to_bf16_rne(float f) {
  uint32_t u = __float_as_uint(f);
  uint32_t r = u + 0x7fffu + ((u >> 16) & 1u);   // rne rounding
  uint32_t q = u | 0x00400000u;                  // quieted NaN
  bool nan = (u & 0x7fffffffu) > 0x7f800000u;
  return (unsigned short)((nan ? q : r) >> 16);
}

__device__ __forceinline__ v16bf cat8(v8bf lo, v8bf hi) {
  return __builtin_shufflevector(lo, hi, 0,1,2,3,4,5,6,7,8,9,10,11,12,13,14,15);
}

// ---------------- small init ----------------
__global__ void init_kernel(int* counts0, float* P_sum) {
  int tid = threadIdx.x;
  if (tid < NEXP) { counts0[tid] = 0; P_sum[tid] = 0.f; }
}

// ---------------- routing: logits -> softmax -> argmax ----------------
__global__ __launch_bounds__(256)
void route_kernel(const float* __restrict__ x, const float* __restrict__ Wsw,
                  float* __restrict__ route_prob, int* __restrict__ routes,
                  float* __restrict__ rpmax, int* __restrict__ counts0,
                  float* __restrict__ P_sum) {
  __shared__ float Ws[NEXP * DMODEL];
  const int tid = threadIdx.x;
  for (int i = tid; i < NEXP * DMODEL; i += 256) Ws[i] = Wsw[i];
  __syncthreads();

  const int t = blockIdx.x * 256 + tid;
  const float* xt = x + (size_t)t * DMODEL;
  float acc[NEXP];
#pragma unroll
  for (int e = 0; e < NEXP; e++) acc[e] = 0.f;
  for (int d = 0; d < DMODEL; d++) {
    float xv = xt[d];
#pragma unroll
    for (int e = 0; e < NEXP; e++) acc[e] = fmaf(xv, Ws[e * DMODEL + d], acc[e]);
  }
  float m = acc[0]; int am = 0;
#pragma unroll
  for (int e = 1; e < NEXP; e++) if (acc[e] > m) { m = acc[e]; am = e; }
  float p[NEXP], s = 0.f;
#pragma unroll
  for (int e = 0; e < NEXP; e++) { p[e] = __expf(acc[e] - m); s += p[e]; }
  float inv = 1.f / s, pm = 0.f;
#pragma unroll
  for (int e = 0; e < NEXP; e++) {
    p[e] *= inv;
    route_prob[(size_t)t * NEXP + e] = p[e];
    atomicAdd(&P_sum[e], p[e]);
    if (p[e] > pm) pm = p[e];
  }
  routes[t] = am;
  rpmax[t] = pm;
  atomicAdd(&counts0[am], 1);
}

// ------- sequential: adaptive capacity, ranking, greedy reassignment -------
__global__ void scan_kernel(const int* __restrict__ routes,
                            const float* __restrict__ route_prob,
                            const int* __restrict__ counts0,
                            int* __restrict__ order, int* __restrict__ flat,
                            int* __restrict__ dropped, int* __restrict__ counts_final) {
  if (threadIdx.x != 0 || blockIdx.x != 0) return;
  int cap[NEXP], starts[NEXP], pos[NEXP], cnt[NEXP], spare[NEXP];
  int run = 0;
  for (int e = 0; e < NEXP; e++) {
    int c0 = counts0[e];
    float dl = ((float)c0 - (float)BASE_CAP) * 0.2f;
    dl = dl < 0.f ? 0.f : (dl > (float)(BASE_CAP / 2) ? (float)(BASE_CAP / 2) : dl);
    cap[e] = BASE_CAP + (int)dl;
    starts[e] = run; pos[e] = run; run += c0;
    int ci = c0 < cap[e] ? c0 : cap[e];
    cnt[e] = ci; spare[e] = cap[e] - ci;
  }
  for (int t = 0; t < T_TOK; t++) { int e = routes[t]; order[pos[e]++] = t; }
  for (int i = 0; i < T_TOK; i++) {
    int t = order[i], e = routes[t], rank = i - starts[e];
    if (rank < cap[e]) { flat[t] = e * CAPMAX + rank; dropped[t] = 0; continue; }
    int has = 0;
    for (int j = 0; j < NEXP; j++) has |= (spare[j] > 0);
    if (has) {
      float best = -3.0e38f; int bj = 0;
      for (int j = 0; j < NEXP; j++)
        if (spare[j] > 0) {
          float cand = route_prob[(size_t)t * NEXP + j] / (1.f + (float)cnt[j]);
          if (cand > best) { best = cand; bj = j; }
        }
      flat[t] = bj * CAPMAX + cnt[bj]; dropped[t] = 0;
      spare[bj]--; cnt[bj]++;
    } else { flat[t] = 0; dropped[t] = 1; }
  }
  for (int e = 0; e < NEXP; e++) counts_final[e] = cnt[e];
}

// ------- scatter tokens into expert buffers, fp32 -> bf16 -------
__global__ __launch_bounds__(256)
void scatter_kernel(const float* __restrict__ x, const int* __restrict__ flat,
                    const int* __restrict__ dropped, unsigned short* __restrict__ ebuf) {
  int idx = blockIdx.x * 256 + threadIdx.x;       // T * D/4 threads
  int t = idx / (DMODEL / 4);
  int c4 = idx % (DMODEL / 4);
  if (dropped[t]) return;
  float4 v = reinterpret_cast<const float4*>(x)[(size_t)t * (DMODEL / 4) + c4];
  ushort4 o;
  o.x = f32_to_bf16_rne(v.x); o.y = f32_to_bf16_rne(v.y);
  o.z = f32_to_bf16_rne(v.z); o.w = f32_to_bf16_rne(v.w);
  reinterpret_cast<ushort4*>(ebuf)[(size_t)flat[t] * (DMODEL / 4) + c4] = o;
}

// ---------------- WMMA bf16 GEMM: O[e] = act(A[e] @ B[e] + bias[e]) ----------------
// A: bf16 [E][M][K] row-major (staged with global_load_async_to_lds_b128).
// B: fp32 [E][K][N] (transposed + converted to bf16 while staging).
// RELU_BF16_OUT: relu + bf16 store (GEMM1 -> H). else: fp32 store (GEMM2 -> ybuf).
#define BM 128
#define BN 128
#define BK 32
#define ASTR 48   // padded LDS row stride (elements); 96B keeps 16B alignment
#define BSTR 48

template <bool RELU_BF16_OUT>
__global__ __launch_bounds__(256)
void moe_gemm_wmma(const __bf16* __restrict__ Abase, const float* __restrict__ Bbase,
                   const float* __restrict__ bias, void* __restrict__ Obase,
                   int M, int N, int K) {
  __shared__ __bf16 As[BM * ASTR];
  __shared__ __bf16 Bs[BN * BSTR];
  const int e = blockIdx.z;
  const int bn = blockIdx.x;
  const int bm = blockIdx.y;
  const int tid = threadIdx.x;
  const int wave = tid >> 5;
  const int lane = tid & 31;
  const int lh = lane & 15;
  const int hi = lane >> 4;
  const int wm = wave & 3;   // 4 wave-slabs along M (32 rows each)
  const int wn = wave >> 2;  // 2 wave-slabs along N (64 cols each)

  const __bf16* Ae = Abase + (size_t)e * M * K + (size_t)bm * BM * K;
  const float*  Be = Bbase + (size_t)e * K * N + (size_t)bn * BN;
  const float*  be = bias  + (size_t)e * N + (size_t)bn * BN;

  v8f acc[2][4];
#pragma unroll
  for (int tm = 0; tm < 2; tm++)
#pragma unroll
    for (int tn = 0; tn < 4; tn++) acc[tm][tn] = (v8f){};

  // B staging decomposition: each thread owns 2 consecutive k rows x 8 n cols.
  const int kk2 = (tid >> 4) * 2;      // 0,2,...,30
  const int ng  = (tid & 15) * 8;      // 0,8,...,120

  for (int kt = 0; kt < K; kt += BK) {
    // --- stage A tile: 128x32 bf16 via async DMA into LDS (no VGPR round-trip) ---
#pragma unroll
    for (int p = 0; p < 2; p++) {
      int ci = tid + p * 256;
      int row = ci >> 2;
      int kc = ci & 3;
      const __bf16* gp = Ae + (size_t)row * K + kt + kc * 8;
      unsigned lds_off = (unsigned)(uintptr_t)(&As[row * ASTR + kc * 8]); // low 32b = LDS offset
      asm volatile("global_load_async_to_lds_b128 %0, %1, off"
                   :: "v"(lds_off), "v"(gp) : "memory");
    }
    // --- stage B tile: 32(k) x 128(n) fp32 -> bf16, transposed to Bs[n][k] ---
    {
      const float* bp0 = Be + (size_t)(kt + kk2) * N + ng;
      const float* bp1 = bp0 + N;
      float4 a0 = reinterpret_cast<const float4*>(bp0)[0];
      float4 a1 = reinterpret_cast<const float4*>(bp0)[1];
      float4 c0 = reinterpret_cast<const float4*>(bp1)[0];
      float4 c1 = reinterpret_cast<const float4*>(bp1)[1];
      float lo[8] = {a0.x, a0.y, a0.z, a0.w, a1.x, a1.y, a1.z, a1.w};
      float hi8[8] = {c0.x, c0.y, c0.z, c0.w, c1.x, c1.y, c1.z, c1.w};
      unsigned* bsw = reinterpret_cast<unsigned*>(Bs);
#pragma unroll
      for (int i = 0; i < 8; i++) {
        unsigned pk = (unsigned)f32_to_bf16_rne(lo[i]) |
                      ((unsigned)f32_to_bf16_rne(hi8[i]) << 16);
        bsw[((ng + i) * BSTR + kk2) >> 1] = pk;   // two bf16 (k, k+1) per dword
      }
    }
    asm volatile("s_wait_asynccnt 0x0" ::: "memory");
    __syncthreads();

    // --- A fragments: lane<16 holds K{0-7,16-23}, lane>=16 holds K{8-15,24-31} ---
    v16bf afrag[2];
#pragma unroll
    for (int tm = 0; tm < 2; tm++) {
      int ar = (wm * 32 + tm * 16 + lh) * ASTR + hi * 8;
      afrag[tm] = cat8(*reinterpret_cast<const v8bf*>(&As[ar]),
                       *reinterpret_cast<const v8bf*>(&As[ar + 16]));
    }
    // --- B fragments: lane<16 holds K{0-7,16-23}, lane>=16 holds K{8-15,24-31} ---
    v16bf bfrag[4];
#pragma unroll
    for (int tn = 0; tn < 4; tn++) {
      int br = (wn * 64 + tn * 16 + lh) * BSTR + hi * 8;
      bfrag[tn] = cat8(*reinterpret_cast<const v8bf*>(&Bs[br]),
                       *reinterpret_cast<const v8bf*>(&Bs[br + 16]));
    }

#pragma unroll
    for (int tm = 0; tm < 2; tm++)
#pragma unroll
      for (int tn = 0; tn < 4; tn++)
        acc[tm][tn] = __builtin_amdgcn_wmma_f32_16x16x32_bf16(
            false, afrag[tm], false, bfrag[tn], (short)0, acc[tm][tn], false, false);
    __syncthreads();
  }

  // --- epilogue: C/D layout -> VGPR v holds row (v + hi*8), col = lh ---
#pragma unroll
  for (int tm = 0; tm < 2; tm++) {
#pragma unroll
    for (int tn = 0; tn < 4; tn++) {
      int ln = wn * 64 + tn * 16 + lh;
      int gn = bn * BN + ln;
      float bv = be[ln];
#pragma unroll
      for (int v = 0; v < 8; v++) {
        int gm = bm * BM + wm * 32 + tm * 16 + v + hi * 8;
        float val = acc[tm][tn][v] + bv;
        if (RELU_BF16_OUT) {
          val = val > 0.f ? val : 0.f;
          reinterpret_cast<unsigned short*>(Obase)[(size_t)e * M * N + (size_t)gm * N + gn] =
              f32_to_bf16_rne(val);
        } else {
          reinterpret_cast<float*>(Obase)[(size_t)e * M * N + (size_t)gm * N + gn] = val;
        }
      }
    }
  }
}

// ------- gather expert outputs (or passthrough for dropped), scale by p_max -------
__global__ __launch_bounds__(256)
void finalize_kernel(const float* __restrict__ x, const float* __restrict__ ybuf,
                     const int* __restrict__ flat, const int* __restrict__ dropped,
                     const float* __restrict__ rpmax, float* __restrict__ out) {
  int idx = blockIdx.x * 256 + threadIdx.x;       // T * D/4 threads
  int t = idx / (DMODEL / 4);
  int c4 = idx % (DMODEL / 4);
  float s = rpmax[t];
  float4 v;
  if (dropped[t]) v = reinterpret_cast<const float4*>(x)[(size_t)t * (DMODEL / 4) + c4];
  else            v = reinterpret_cast<const float4*>(ybuf)[(size_t)flat[t] * (DMODEL / 4) + c4];
  v.x *= s; v.y *= s; v.z *= s; v.w *= s;
  reinterpret_cast<float4*>(out)[idx] = v;
}

// ------- aux losses: load-balance + sim-balance -------
__global__ void loss_kernel(const float* __restrict__ Wsw, const int* __restrict__ counts_final,
                            const float* __restrict__ P_sum, float* __restrict__ out_loss) {
  __shared__ float red[64];
  int tid = threadIdx.x;
  int i = tid >> 3, j = tid & 7;
  float dot = 0.f;
  for (int d = 0; d < DMODEL; d++) dot = fmaf(Wsw[i * DMODEL + d], Wsw[j * DMODEL + d], dot);
  red[tid] = (i != j) ? dot * dot : 0.f;
  __syncthreads();
  if (tid == 0) {
    float simbal = 0.f;
    for (int k = 0; k < 64; k++) simbal += red[k];
    simbal *= 0.001f;
    float lb = 0.f;
    for (int e = 0; e < NEXP; e++)
      lb += ((float)counts_final[e] / (float)T_TOK) * (P_sum[e] / (float)T_TOK);
    lb *= 0.01f * (float)NEXP;
    *out_loss = lb + simbal;
  }
}

// ---------------- host-side launch ----------------
extern "C" void kernel_launch(void* const* d_in, const int* in_sizes, int n_in,
                              void* d_out, int out_size, void* d_ws, size_t ws_size,
                              hipStream_t stream) {
  (void)in_sizes; (void)n_in; (void)out_size; (void)ws_size;
  const float* x   = (const float*)d_in[0];
  const float* Wsw = (const float*)d_in[1];
  const float* W1  = (const float*)d_in[2];
  const float* b1  = (const float*)d_in[3];
  const float* W2  = (const float*)d_in[4];
  const float* b2  = (const float*)d_in[5];
  float* out = (float*)d_out;

  char* ws = (char*)d_ws;
  size_t off = 0;
#define TAKE(var, type, count) \
  type* var = (type*)(ws + off); off += (((size_t)(count) * sizeof(type)) + 255) & ~(size_t)255;

  TAKE(route_prob, float, (size_t)T_TOK * NEXP)
  TAKE(routes, int, T_TOK)
  TAKE(rpmax, float, T_TOK)
  TAKE(counts0, int, NEXP)
  TAKE(P_sum, float, NEXP)
  TAKE(order, int, T_TOK)
  TAKE(flat, int, T_TOK)
  TAKE(dropped, int, T_TOK)
  TAKE(counts_final, int, NEXP)
  TAKE(ebuf, unsigned short, (size_t)NEXP * CAPMAX * DMODEL)           // bf16 [E][C][D]
  TAKE(Hbuf, unsigned short, (size_t)NEXP * CAPMAX * FDIM)             // bf16 [E][C][F]
  TAKE(ybuf, float, (size_t)NEXP * CAPMAX * DMODEL)                    // f32  [E][C][D]
#undef TAKE

  init_kernel<<<1, 32, 0, stream>>>(counts0, P_sum);
  route_kernel<<<T_TOK / 256, 256, 0, stream>>>(x, Wsw, route_prob, routes, rpmax, counts0, P_sum);
  scan_kernel<<<1, 32, 0, stream>>>(routes, route_prob, counts0, order, flat, dropped, counts_final);
  scatter_kernel<<<(T_TOK * (DMODEL / 4)) / 256, 256, 0, stream>>>(x, flat, dropped, ebuf);

  // GEMM1: H = relu(ebuf @ W1 + b1)   [E] 768x768 @ 768x3072
  dim3 g1(FDIM / BN, CAPMAX / BM, NEXP);
  moe_gemm_wmma<true><<<g1, 256, 0, stream>>>(
      reinterpret_cast<const __bf16*>(ebuf), W1, b1, (void*)Hbuf, CAPMAX, FDIM, DMODEL);

  // GEMM2: y = H @ W2 + b2            [E] 768x3072 @ 3072x768
  dim3 g2(DMODEL / BN, CAPMAX / BM, NEXP);
  moe_gemm_wmma<false><<<g2, 256, 0, stream>>>(
      reinterpret_cast<const __bf16*>(Hbuf), W2, b2, (void*)ybuf, CAPMAX, DMODEL, FDIM);

  finalize_kernel<<<(T_TOK * (DMODEL / 4)) / 256, 256, 0, stream>>>(x, ybuf, flat, dropped, rpmax, out);
  loss_kernel<<<1, 64, 0, stream>>>(Wsw, counts_final, P_sum, out + (size_t)T_TOK * DMODEL);
}
